// model_congen_32040456028953
// MI455X (gfx1250) — compile-verified, matrix operand order checked by reading, not run on hardware
//
#include <hip/hip_runtime.h>

#define NB 256   // batch
#define NH 1024  // hidden

typedef __bf16 v16bf __attribute__((ext_vector_type(16)));
typedef __bf16 v8bf  __attribute__((ext_vector_type(8)));
typedef float  v8f   __attribute__((ext_vector_type(8)));
typedef float  v4f   __attribute__((ext_vector_type(4)));

__device__ __forceinline__ float sigm_(float x) { return 1.0f / (1.0f + expf(-x)); }

// ---------------------------------------------------------------------------
// WMMA fragment loads (16x16x32 bf16). Per CDNA5 ISA 16-bit A layout:
// lane l holds row (l&15); element e -> K = (l>>4)*8 + e + 8*(e>=8),
// i.e. two contiguous 8-element K-runs at khi and khi+16.
// ---------------------------------------------------------------------------
__device__ __forceinline__ v16bf load_a_frag(const __bf16* A, int lda, int m, int k, int lane) {
  int row = m + (lane & 15);
  int khi = (lane >> 4) * 8;
  const __bf16* p = A + (size_t)row * lda + (k + khi);
  v8bf lo = *(const v8bf*)p;          // K = khi .. khi+7
  v8bf hi = *(const v8bf*)(p + 16);   // K = khi+16 .. khi+23
  v16bf r;
#pragma unroll
  for (int i = 0; i < 8; ++i) { r[i] = lo[i]; r[i + 8] = hi[i]; }
  return r;
}

// B fragment from the LDS-staged tile (bf16 [64 rows(n)][32 (k)]).
__device__ __forceinline__ v16bf lds_b_frag(const __bf16* buf, int nrel, int lane) {
  int row = nrel + (lane & 15);
  int khi = (lane >> 4) * 8;
  const __bf16* p = buf + row * 32 + khi;
  v8bf lo = *(const v8bf*)p;
  v8bf hi = *(const v8bf*)(p + 16);
  v16bf r;
#pragma unroll
  for (int i = 0; i < 8; ++i) { r[i] = lo[i]; r[i + 8] = hi[i]; }
  return r;
}

// Cooperative stage of one W tile [64 n][32 k] fp32 -> bf16 LDS.
// 256 threads: thread t loads row t>>2, 8 floats at col (t&3)*8 (2x b128),
// converts, one ds_store_b128. Each weight byte crosses HBM exactly once.
__device__ __forceinline__ void stage_w_tile(const float* __restrict__ W, int ldw,
                                             int n0, int k, int nmax,
                                             __bf16* buf, int tid) {
  int r = tid >> 2;              // 0..63
  int c = (tid & 3) * 8;         // 0,8,16,24
  int col = n0 + r;
  col = (col > nmax) ? nmax : col;
  const float* p = W + (size_t)col * ldw + (k + c);
  v4f f0 = *(const v4f*)p;
  v4f f1 = *(const v4f*)(p + 4);
  v8bf h;
#pragma unroll
  for (int i = 0; i < 4; ++i) { h[i] = (__bf16)f0[i]; h[i + 4] = (__bf16)f1[i]; }
  *(v8bf*)(buf + r * 32 + c) = h;
}

// ---------------------------------------------------------------------------
// C[256,N] = A[256,Ka+Kb] * [W0 | W1]^T + bias0 + bias1   (W row-major [N,K])
// Block: 256 thr = 8 waves; block tile 256(M) x 64(N); wave tile 64x32
// (4x2 WMMA 16x16x32 tiles). W staged via LDS (double-buffered, 1 barrier
// per K-step). blockIdx.z = direction.
// ---------------------------------------------------------------------------
__global__ __launch_bounds__(256) void gemm_wmma_bf16(
    const __bf16* __restrict__ Abase, int lda, long aStride,
    const float* __restrict__ W0d0, const float* __restrict__ W1d0,
    const float* __restrict__ W0d1, const float* __restrict__ W1d1,
    int K0, int K1,
    const float* __restrict__ b0d0, const float* __restrict__ b1d0,
    const float* __restrict__ b0d1, const float* __restrict__ b1d1,
    float* __restrict__ Cbase, int ldc, long cStride, int N) {
  __shared__ __align__(16) __bf16 bufs[2][64 * 32];

  int dir = blockIdx.z;
  const __bf16* A = Abase + (size_t)dir * aStride;
  float* C = Cbase + (size_t)dir * cStride;
  const float* Wseg[2] = { dir ? W0d1 : W0d0, dir ? W1d1 : W1d0 };
  const float* bias0 = dir ? b0d1 : b0d0;
  const float* bias1 = dir ? b1d1 : b1d0;
  int Kseg[2] = { K0, K1 };

  int tid = threadIdx.x;
  int lane = tid & 31;
  int wave = tid >> 5;
  int m0 = (wave & 3) * 64;                       // 4 M-waves cover M=256
  int nTile0 = blockIdx.x * 64;                   // block's N range
  int nrel = (wave >> 2) * 32;                    // 2 N-waves cover 64
  int n0 = nTile0 + nrel;

  v8f acc[4][2] = {};

  int kA = 0;
  for (int seg = 0; seg < 2; ++seg) {
    const float* W = Wseg[seg];
    int K = Kseg[seg];
    if (W && K > 0) {
      int nsteps = K / 32;
      stage_w_tile(W, K, nTile0, 0, N - 1, bufs[0], tid);
      __syncthreads();
      for (int s = 0; s < nsteps; ++s) {
        int k = s * 32;
        if (s + 1 < nsteps)
          stage_w_tile(W, K, nTile0, k + 32, N - 1, bufs[(s + 1) & 1], tid);
        const __bf16* buf = bufs[s & 1];
        v16bf b0 = lds_b_frag(buf, nrel, lane);
        v16bf b1 = lds_b_frag(buf, nrel + 16, lane);
        v16bf a0 = load_a_frag(A, lda, m0,      kA + k, lane);
        v16bf a1 = load_a_frag(A, lda, m0 + 16, kA + k, lane);
        v16bf a2 = load_a_frag(A, lda, m0 + 32, kA + k, lane);
        v16bf a3 = load_a_frag(A, lda, m0 + 48, kA + k, lane);
#define WMMA_ACC(I, AV, J, BV) \
        acc[I][J] = __builtin_amdgcn_wmma_f32_16x16x32_bf16(false, AV, false, BV, (short)0, acc[I][J], false, false)
        WMMA_ACC(0, a0, 0, b0); WMMA_ACC(0, a0, 1, b1);
        WMMA_ACC(1, a1, 0, b0); WMMA_ACC(1, a1, 1, b1);
        WMMA_ACC(2, a2, 0, b0); WMMA_ACC(2, a2, 1, b1);
        WMMA_ACC(3, a3, 0, b0); WMMA_ACC(3, a3, 1, b1);
#undef WMMA_ACC
        __syncthreads();
      }
    }
    kA += K;
  }

  // Epilogue: C/D layout -> lane l, vgpr v : M = (l>>4)*8 + v, N = l&15
  int rbase = (lane >> 4) * 8;
  int nc = lane & 15;
#pragma unroll
  for (int j = 0; j < 2; ++j) {
    int n = n0 + j * 16 + nc;
    if (n < N) {
      float bv = (bias0 ? bias0[n] : 0.0f) + (bias1 ? bias1[n] : 0.0f);
#pragma unroll
      for (int i = 0; i < 4; ++i) {
        int mb = m0 + i * 16 + rbase;
#pragma unroll
        for (int v = 0; v < 8; ++v)
          C[(size_t)(mb + v) * ldc + n] = acc[i][j][v] + bv;
      }
    }
  }
}

// ---------------------------------------------------------------------------
// x1[dir][b][0:3]=inp, [3:64]=old_w, [64:1088]=h1[dir]   (bf16 staging)
// ---------------------------------------------------------------------------
__global__ __launch_bounds__(256) void pack_x1(
    const float* __restrict__ inp, const float* __restrict__ old_w,
    const float* __restrict__ h1, __bf16* __restrict__ x1) {
  int idx = blockIdx.x * blockDim.x + threadIdx.x;
  if (idx >= 2 * NB * 1088) return;
  int col = idx % 1088;
  int b = (idx / 1088) % NB;
  int dir = idx / (1088 * NB);
  float v;
  if (col < 3)       v = inp[b * 3 + col];
  else if (col < 64) v = old_w[b * 61 + (col - 3)];
  else               v = h1[((size_t)dir * NB + b) * NH + (col - 64)];
  x1[idx] = (__bf16)v;
}

// ---------------------------------------------------------------------------
// LSTM cell pointwise: gates [2][B][4096] (i,f,g,o) -> h_out/c_out (fp32),
// plus bf16 copies of h into up to 3 downstream GEMM input slabs.
// ---------------------------------------------------------------------------
__global__ __launch_bounds__(256) void lstm_act(
    const float* __restrict__ gates, const float* __restrict__ c_in,
    float* __restrict__ h_out, float* __restrict__ c_out,
    __bf16* __restrict__ s0, int ld0, int base0,
    __bf16* __restrict__ s1, int ld1, int base1,
    __bf16* __restrict__ s2, int ld2, int base2) {
  int idx = blockIdx.x * blockDim.x + threadIdx.x;   // 2*B*H
  if (idx >= 2 * NB * NH) return;
  int j = idx & (NH - 1);
  int b = (idx >> 10) & (NB - 1);
  int dir = idx >> 18;
  const float* g = gates + ((size_t)dir * NB + b) * 4096;
  float gi = g[j], gf = g[1024 + j], gg = g[2048 + j], go = g[3072 + j];
  float co = c_in[idx];
  float cn = sigm_(gf) * co + sigm_(gi) * tanhf(gg);
  float hn = sigm_(go) * tanhf(cn);
  c_out[idx] = cn;
  h_out[idx] = hn;
  int col = dir * NH + j;
  __bf16 hb = (__bf16)hn;
  if (s0) s0[(size_t)b * ld0 + base0 + col] = hb;
  if (s1) s1[(size_t)b * ld1 + base1 + col] = hb;
  if (s2) s2[(size_t)b * ld2 + base2 + col] = hb;
}

// ---------------------------------------------------------------------------
// Gaussian mixture attention window: one block per batch row.
// ---------------------------------------------------------------------------
__global__ __launch_bounds__(64) void window_kernel(
    const float* __restrict__ abk, const float* __restrict__ old_k,
    const float* __restrict__ char_vec, const int* __restrict__ text_len,
    float* __restrict__ phi_out, float* __restrict__ win_out,
    float* __restrict__ kt_out) {
  __shared__ float sa[10], sb[10], sk[10], sphi[64];
  int b = blockIdx.x, t = threadIdx.x;
  const float* y = abk + b * 30;
  if (t < 10) {
    sa[t] = expf(y[t]);
    sb[t] = expf(y[10 + t]);
    float kv = old_k[b * 10 + t] + expf(y[20 + t]);
    sk[t] = kv;
    kt_out[b * 10 + t] = kv;
  }
  __syncthreads();
  float scale = 64.0f / (float)(*text_len);
  float uu = (float)(t + 1);
  float p = 0.0f;
#pragma unroll
  for (int k = 0; k < 10; ++k) {
    float d = sk[k] - uu;
    p += sa[k] * expf(-sb[k] * d * d);
  }
  p *= scale;
  sphi[t] = p;
  phi_out[b * 64 + t] = p;
  __syncthreads();
  if (t < 61) {
    float acc = 0.0f;
    const float* cv = char_vec + (size_t)b * 64 * 61 + t;
#pragma unroll 8
    for (int u = 0; u < 64; ++u) acc += sphi[u] * cv[u * 61];
    win_out[b * 61 + t] = acc;
  }
}

// x2[dir][b][2048:2051]=inp, [2051:2112]=win, [2112:3136]=h2[dir]
__global__ __launch_bounds__(256) void pack_x2_tail(
    const float* __restrict__ inp, const float* __restrict__ win,
    const float* __restrict__ h2, __bf16* __restrict__ x2f,
    __bf16* __restrict__ x2b) {
  int idx = blockIdx.x * blockDim.x + threadIdx.x;
  if (idx >= 2 * NB * 1088) return;
  int col = idx % 1088;
  int b = (idx / 1088) % NB;
  int dir = idx / (1088 * NB);
  float v;
  if (col < 3)       v = inp[b * 3 + col];
  else if (col < 64) v = win[b * 61 + (col - 3)];
  else               v = h2[((size_t)dir * NB + b) * NH + (col - 64)];
  __bf16* dst = dir ? x2b : x2f;
  dst[(size_t)b * 3136 + 2048 + col] = (__bf16)v;
}

// MDN head epilogue: e, softmax(pi), mu, exp(s), tanh(rho). One wave per b.
__global__ __launch_bounds__(32) void mdn_epilogue(
    const float* __restrict__ y, float* __restrict__ e, float* __restrict__ pi,
    float* __restrict__ mu1, float* __restrict__ mu2, float* __restrict__ s1,
    float* __restrict__ s2, float* __restrict__ rho) {
  int b = blockIdx.x, t = threadIdx.x;
  const float* yb = y + b * 121;
  if (t == 0) e[b] = sigm_(yb[0]);
  float v = (t < 20) ? yb[1 + t] : -1e30f;
  float m = v;
  for (int o = 16; o > 0; o >>= 1) m = fmaxf(m, __shfl_xor(m, o));
  float ex = (t < 20) ? expf(v - m) : 0.0f;
  float s = ex;
  for (int o = 16; o > 0; o >>= 1) s += __shfl_xor(s, o);
  if (t < 20) {
    pi[b * 20 + t] = ex / s;
    mu1[b * 20 + t] = yb[21 + t];
    mu2[b * 20 + t] = yb[41 + t];
    s1[b * 20 + t] = expf(yb[61 + t]);
    s2[b * 20 + t] = expf(yb[81 + t]);
    rho[b * 20 + t] = tanhf(yb[101 + t]);
  }
}

// ---------------------------------------------------------------------------
extern "C" void kernel_launch(void* const* d_in, const int* in_sizes, int n_in,
                              void* d_out, int out_size, void* d_ws, size_t ws_size,
                              hipStream_t stream) {
  (void)in_sizes; (void)n_in; (void)out_size; (void)ws_size;
  const float* inp      = (const float*)d_in[0];
  const float* char_vec = (const float*)d_in[1];
  const float* old_k    = (const float*)d_in[2];
  const float* old_w    = (const float*)d_in[3];
  const float* h1       = (const float*)d_in[4];
  const float* c1       = (const float*)d_in[5];
  const float* h2       = (const float*)d_in[6];
  const float* c2       = (const float*)d_in[7];
  const int*   text_len = (const int*)d_in[8];
  const float* Wih1f = (const float*)d_in[9];
  const float* Whh1f = (const float*)d_in[10];
  const float* bih1f = (const float*)d_in[11];
  const float* bhh1f = (const float*)d_in[12];
  const float* Wih1b = (const float*)d_in[13];
  const float* Whh1b = (const float*)d_in[14];
  const float* bih1b = (const float*)d_in[15];
  const float* bhh1b = (const float*)d_in[16];
  const float* Wih2f = (const float*)d_in[17];
  const float* Whh2f = (const float*)d_in[18];
  const float* bih2f = (const float*)d_in[19];
  const float* bhh2f = (const float*)d_in[20];
  const float* Wih2b = (const float*)d_in[21];
  const float* Whh2b = (const float*)d_in[22];
  const float* bih2b = (const float*)d_in[23];
  const float* bhh2b = (const float*)d_in[24];
  const float* W_win = (const float*)d_in[25];
  const float* b_win = (const float*)d_in[26];
  const float* W_mdn = (const float*)d_in[27];
  const float* b_mdn = (const float*)d_in[28];

  // workspace carve (256B aligned)
  char* w = (char*)d_ws;
  auto carve = [&](size_t bytes) { char* p = w; w += (bytes + 255) & ~(size_t)255; return p; };
  __bf16* xcat1  = (__bf16*)carve((size_t)2 * NB * 1088 * 2);  // [2][B][1088]
  float*  gates  = (float*) carve((size_t)2 * NB * 4096 * 4);  // reused layer1/layer2
  __bf16* xcat2  = (__bf16*)carve((size_t)2 * NB * 3136 * 2);  // [2][B][3136]
  __bf16* mdn_in = (__bf16*)carve((size_t)NB * 4096 * 2);      // [B][out1|out2]
  float*  abkb   = (float*) carve((size_t)NB * 30 * 4);
  float*  ymdn   = (float*) carve((size_t)NB * 121 * 4);
  __bf16* x2b = xcat2 + (size_t)NB * 3136;

  // d_out layout (reference return order)
  float* out = (float*)d_out;
  float* o_e   = out;             // [B,1]
  float* o_pi  = o_e  + 256;      // [B,20]
  float* o_mu1 = o_pi + 5120;
  float* o_mu2 = o_mu1 + 5120;
  float* o_s1  = o_mu2 + 5120;
  float* o_s2  = o_s1 + 5120;
  float* o_rho = o_s2 + 5120;
  float* o_phi = o_rho + 5120;    // [B,64]
  float* o_win = o_phi + 16384;   // [B,61]
  float* o_kt  = o_win + 15616;   // [B,10]
  float* o_h1n = o_kt  + 2560;    // [2,B,H]
  float* o_c1n = o_h1n + 524288;
  float* o_h2n = o_c1n + 524288;
  float* o_c2n = o_h2n + 524288;

  dim3 blk(256);

  // 1) stage x1 = [inp | old_w | h1[dir]] as bf16
  pack_x1<<<dim3((2 * NB * 1088 + 255) / 256), blk, 0, stream>>>(inp, old_w, h1, xcat1);

  // 2) gates1 = x1 @ [Wih1|Whh1]^T + bih1 + bhh1   (both directions)
  gemm_wmma_bf16<<<dim3(64, 1, 2), blk, 0, stream>>>(
      xcat1, 1088, (long)NB * 1088,
      Wih1f, Whh1f, Wih1b, Whh1b, 64, 1024,
      bih1f, bhh1f, bih1b, bhh1b,
      gates, 4096, (long)NB * 4096, 4096);

  // 3) LSTM1 cell -> h1n/c1n; h1n(bf16) into mdn_in[:,0:2048], xcat2{f,b}[:,0:2048]
  lstm_act<<<dim3(2048), blk, 0, stream>>>(gates, c1, o_h1n, o_c1n,
                                           mdn_in, 4096, 0,
                                           xcat2, 3136, 0,
                                           x2b, 3136, 0);

  // 4) abk = out1 @ W_win^T + b_win   (out1 = mdn_in[:,0:2048])
  gemm_wmma_bf16<<<dim3(1, 1, 1), blk, 0, stream>>>(
      mdn_in, 4096, 0,
      W_win, nullptr, W_win, nullptr, 2048, 0,
      b_win, nullptr, b_win, nullptr,
      abkb, 30, 0, 30);

  // 5) attention window: phi, win, k_t
  window_kernel<<<dim3(NB), dim3(64), 0, stream>>>(abkb, old_k, char_vec, text_len,
                                                   o_phi, o_win, o_kt);

  // 6) finish x2 tails: [inp | win | h2[dir]]
  pack_x2_tail<<<dim3((2 * NB * 1088 + 255) / 256), blk, 0, stream>>>(inp, o_win, h2,
                                                                      xcat2, x2b);

  // 7) gates2 = x2 @ [Wih2|Whh2]^T + bih2 + bhh2
  gemm_wmma_bf16<<<dim3(64, 1, 2), blk, 0, stream>>>(
      xcat2, 3136, (long)NB * 3136,
      Wih2f, Whh2f, Wih2b, Whh2b, 2112, 1024,
      bih2f, bhh2f, bih2b, bhh2b,
      gates, 4096, (long)NB * 4096, 4096);

  // 8) LSTM2 cell -> h2n/c2n; h2n(bf16) into mdn_in[:,2048:4096]
  lstm_act<<<dim3(2048), blk, 0, stream>>>(gates, c2, o_h2n, o_c2n,
                                           mdn_in, 4096, 2048,
                                           nullptr, 0, 0,
                                           nullptr, 0, 0);

  // 9) y = [out1|out2] @ W_mdn^T + b_mdn
  gemm_wmma_bf16<<<dim3(2, 1, 1), blk, 0, stream>>>(
      mdn_in, 4096, 0,
      W_mdn, nullptr, W_mdn, nullptr, 4096, 0,
      b_mdn, nullptr, b_mdn, nullptr,
      ymdn, 121, 0, 121);

  // 10) MDN head epilogue
  mdn_epilogue<<<dim3(NB), dim3(32), 0, stream>>>(ymdn, o_e, o_pi, o_mu1, o_mu2,
                                                  o_s1, o_s2, o_rho);
}
